// LlamaChunkAttention_36481452212595
// MI455X (gfx1250) — matrix operand order repaired and checked by reading, not compiled, not run on hardware
//
#include <hip/hip_runtime.h>

#define HIDDEN   4096
#define N_HEADS  32
#define N_KV     8
#define HEAD_DIM 128
#define BB       2
#define SS       2048
#define TOKENS   (BB * SS)          // 4096
#define KV_DIM   (N_KV * HEAD_DIM)  // 1024
#define SCALE    0.08838834764831845f  // 1/sqrt(128)
#define NEG_BIG  (-1.0e30f)
#define LN_ROPE  9.210340371976184f    // ln(10000)
#define TILE_K   32

typedef __attribute__((ext_vector_type(16))) __bf16 v16bf;
typedef __attribute__((ext_vector_type(8)))  __bf16 v8bf;
typedef __attribute__((ext_vector_type(8)))  float  v8f;
typedef __attribute__((ext_vector_type(4)))  unsigned v4u;
typedef __attribute__((ext_vector_type(8)))  int    v8i;
typedef __attribute__((ext_vector_type(4)))  int    v4i;

union Frag { v16bf v; v8bf h[2]; };

// ---- CDNA5 feature detection (graceful degradation keeps it compiling) ----
#if defined(__has_builtin)
#if __has_builtin(__builtin_amdgcn_global_load_async_to_lds_b128)
#define HAS_ASYNC_LDS 1
#endif
#if __has_builtin(__builtin_amdgcn_tensor_load_to_lds)
#define HAS_TDM 1
#endif
#if __has_builtin(__builtin_amdgcn_s_wait_asynccnt)
#define HAS_WAIT_ASYNC 1
#endif
#if __has_builtin(__builtin_amdgcn_s_wait_tensorcnt)
#define HAS_WAIT_TENSOR 1
#endif
#endif

static __device__ __forceinline__ v8f wmma_bf16(v16bf a, v16bf b, v8f c) {
    return __builtin_amdgcn_wmma_f32_16x16x32_bf16(false, a, false, b, (short)0, c,
                                                   false, false);
}

static __device__ __forceinline__ float half16_max(float x) {
    x = fmaxf(x, __shfl_xor(x, 1, 32));
    x = fmaxf(x, __shfl_xor(x, 2, 32));
    x = fmaxf(x, __shfl_xor(x, 4, 32));
    x = fmaxf(x, __shfl_xor(x, 8, 32));
    return x;
}
static __device__ __forceinline__ float half16_sum(float x) {
    x += __shfl_xor(x, 1, 32);
    x += __shfl_xor(x, 2, 32);
    x += __shfl_xor(x, 4, 32);
    x += __shfl_xor(x, 8, 32);
    return x;
}

// 16-byte global->LDS copy: async engine if available, else through VGPRs.
static __device__ __forceinline__ void copy16B_to_lds(__bf16* l, const __bf16* g) {
#if HAS_ASYNC_LDS
    __builtin_amdgcn_global_load_async_to_lds_b128(
        (__attribute__((address_space(1))) v4i*)(__bf16*)g,
        (__attribute__((address_space(3))) v4i*)l, 0, 0);
#else
    *(v8bf*)l = *(const v8bf*)g;
#endif
}

static __device__ __forceinline__ void wait_async_self() {
#if HAS_WAIT_ASYNC
    __builtin_amdgcn_s_wait_asynccnt(0);
#else
    asm volatile("s_wait_asynccnt 0" ::: "memory");
#endif
}

static __device__ __forceinline__ void wait_tensor_self() {
#if HAS_WAIT_TENSOR
    __builtin_amdgcn_s_wait_tensorcnt(0);
#else
    asm volatile("s_wait_tensorcnt 0" ::: "memory");
#endif
}

// Stage a 64-row x 32-col bf16 tile (row stride K elems) into LDS [64][32],
// cooperatively across 128 threads (2 x b128 per thread).
static __device__ __forceinline__ void stage_tile(__bf16* dst, const __bf16* src,
                                                  int K, int tid) {
#pragma unroll
    for (int ch = tid; ch < 256; ch += 128) {
        const int row = ch >> 2;
        const int kc = (ch & 3) * 8;
        copy16B_to_lds(dst + row * 32 + kc, src + (size_t)row * K + kc);
    }
}

#if HAS_TDM
// One TDM descriptor: 2-D tile, 64 rows x 32 elems of 2-byte data, row stride
// rowStrideElems, into contiguous LDS [64][32]. Issued by a single wave.
static __device__ __forceinline__ void tdm_stage_tile(__bf16* lds, const __bf16* g,
                                                      int rowStrideElems,
                                                      int tensorRows) {
    const unsigned long long ga = (unsigned long long)(size_t)g;
    const unsigned lds_off =
        (unsigned)(size_t)(__attribute__((address_space(3))) __bf16*)lds;
    const unsigned d0 = (unsigned)rowStrideElems;  // tensor_dim0 (>= 32)
    const unsigned d1 = (unsigned)tensorRows;      // tensor_dim1 (>= 64)

    v4u g0 = {0u, 0u, 0u, 0u};
    g0[0] = 1u;                                        // count=1, user mode
    g0[1] = lds_off;                                   // lds_addr
    g0[2] = (unsigned)ga;                              // global_addr[31:0]
    g0[3] = (unsigned)((ga >> 32) & 0x1FFFFFFull) | (2u << 30);  // [56:32] | type=2

    v8i g1 = {0, 0, 0, 0, 0, 0, 0, 0};
    g1[0] = (int)(1u << 16);                           // workgroup_mask=0, data_size=1 (2B)
    g1[1] = (int)((d0 & 0xFFFFu) << 16);               // tensor_dim0[15:0]
    g1[2] = (int)(((d0 >> 16) & 0xFFFFu) | ((d1 & 0xFFFFu) << 16));
    g1[3] = (int)(((d1 >> 16) & 0xFFFFu) | (32u << 16));  // tile_dim0 = 32
    g1[4] = 64;                                        // tile_dim1 = 64, tile_dim2 = 0
    g1[5] = (int)d0;                                   // tensor_dim0_stride[31:0]
    g1[6] = 0;
    g1[7] = 0;

    const v4i z4 = {0, 0, 0, 0};
    const v8i z8 = {0, 0, 0, 0, 0, 0, 0, 0};
    __builtin_amdgcn_tensor_load_to_lds(g0, g1, z4, z4, z8, 0);
}
#endif

// ---------------------------------------------------------------- convert
__global__ void cvt_f32_to_bf16(const float* __restrict__ src,
                                __bf16* __restrict__ dst, long n) {
    long i = (long)blockIdx.x * blockDim.x + threadIdx.x;
    long stride = (long)gridDim.x * blockDim.x;
    for (; i < n; i += stride) dst[i] = (__bf16)src[i];
}

// src[K][N] (f32) -> dst[N][K] (bf16)
__global__ void transpose_cvt_f32_to_bf16(const float* __restrict__ src,
                                          __bf16* __restrict__ dst, int K, int N) {
    long total = (long)K * N;
    long stride = (long)gridDim.x * blockDim.x;
    for (long i = (long)blockIdx.x * blockDim.x + threadIdx.x; i < total; i += stride) {
        int k = (int)(i / N);
        int n = (int)(i % N);
        dst[(size_t)n * K + k] = (__bf16)src[i];
    }
}

// ---------------------------------------------------------------- GEMM
// C[M][N] = A[M][K] @ W[K][N], Bt = W^T stored [N][K] bf16.
// Double-buffered LDS pipeline: A tile via async-to-LDS (per-lane b128),
// B tile via one TDM descriptor (wave 0). 4 waves x 16 rows, 64x64 C tile.
// mode 0: bf16 C[m][n];  mode 1: bf16 C[n][m];  mode 2: f32 C[m][n].
__global__ __launch_bounds__(128)
void gemm_bf16_wmma(const __bf16* __restrict__ A, const __bf16* __restrict__ Bt,
                    void* __restrict__ Cout, int M, int N, int K, int mode) {
    __shared__ __attribute__((aligned(64))) __bf16 aS[2][64 * 32];
    __shared__ __attribute__((aligned(64))) __bf16 bS[2][64 * 32];

    const int tid = threadIdx.x;
    const int lane = tid & 31;
    const int wave = tid >> 5;
    const int hi = lane >> 4;
    const int l15 = lane & 15;
    const int mBase = blockIdx.y * 64;
    const int nBase = blockIdx.x * 64;

    const __bf16* aTile0 = A + (size_t)mBase * K;
    const __bf16* bTile0 = Bt + (size_t)nBase * K;

    v8f acc[4] = {};

    // prologue: fill buffer 0
    stage_tile(&aS[0][0], aTile0, K, tid);
#if HAS_TDM
    if (wave == 0) tdm_stage_tile(&bS[0][0], bTile0, K, N);
#else
    stage_tile(&bS[0][0], bTile0, K, tid);
#endif

    int buf = 0;
    for (int kb = 0; kb < K; kb += TILE_K) {
        wait_async_self();
#if HAS_TDM
        if (wave == 0) wait_tensor_self();
#endif
        __syncthreads();

        if (kb + TILE_K < K) {
            stage_tile(&aS[buf ^ 1][0], aTile0 + kb + TILE_K, K, tid);
#if HAS_TDM
            if (wave == 0) tdm_stage_tile(&bS[buf ^ 1][0], bTile0 + kb + TILE_K, K, N);
#else
            stage_tile(&bS[buf ^ 1][0], bTile0 + kb + TILE_K, K, tid);
#endif
        }

        // Load ALL fragments for this k-step first, then issue the WMMA chain
        // back-to-back (single dscnt wait instead of one per WMMA).
        Frag a;
        const __bf16* aRow = &aS[buf][(wave * 16 + l15) * 32];
        a.h[0] = *(const v8bf*)(aRow + 16 * hi);
        a.h[1] = *(const v8bf*)(aRow + 16 * hi + 8);
        v16bf bv[4];
#pragma unroll
        for (int j = 0; j < 4; ++j)
            bv[j] = *(const v16bf*)&bS[buf][(j * 16 + l15) * 32 + 16 * hi];
#pragma unroll
        for (int j = 0; j < 4; ++j) acc[j] = wmma_bf16(a.v, bv[j], acc[j]);
        buf ^= 1;
    }

#pragma unroll
    for (int j = 0; j < 4; ++j) {
#pragma unroll
        for (int r = 0; r < 8; ++r) {
            const int mm = mBase + wave * 16 + r + 8 * hi;
            const int nn = nBase + j * 16 + l15;
            if (mode == 0)
                ((__bf16*)Cout)[(size_t)mm * N + nn] = (__bf16)acc[j][r];
            else if (mode == 1)
                ((__bf16*)Cout)[(size_t)nn * M + mm] = (__bf16)acc[j][r];
            else
                ((float*)Cout)[(size_t)mm * N + nn] = acc[j][r];
        }
    }
}

// ---------------------------------------------------------------- RoPE (Q and K, in place, bf16)
__global__ __launch_bounds__(256)
void rope_bf16(__bf16* __restrict__ Q, __bf16* __restrict__ Kc,
               const int* __restrict__ pos) {
    const int t = blockIdx.x;  // global token b*S+s
    const float p = (float)pos[t];
    const int total = (N_HEADS + N_KV) * 64;
    for (int idx = threadIdx.x; idx < total; idx += blockDim.x) {
        const int head = idx >> 6;
        const int j = idx & 63;
        const float inv = __expf(-(float)j * (LN_ROPE / 64.0f));  // 10000^(-j/64)
        const float ang = p * inv;
        const float c = __cosf(ang), s = __sinf(ang);
        __bf16* ptr = (head < N_HEADS)
                          ? (Q + (size_t)t * HIDDEN + head * HEAD_DIM)
                          : (Kc + (size_t)t * KV_DIM + (head - N_HEADS) * HEAD_DIM);
        const float x1 = (float)ptr[j];
        const float x2 = (float)ptr[j + 64];
        ptr[j]      = (__bf16)(x1 * c - x2 * s);
        ptr[j + 64] = (__bf16)(x2 * c + x1 * s);
    }
}

// ---------------------------------------------------------------- Flash attention
// One wave per (b, h, 16-query tile). Q [tok][4096], K [tok][1024] (natural),
// Vt [1024 dims][4096 tok] (transposed). Out Ob [tok][4096] bf16.
__global__ __launch_bounds__(32)
void flash_wmma(const __bf16* __restrict__ Qb, const __bf16* __restrict__ Kb,
                const __bf16* __restrict__ Vt, __bf16* __restrict__ Ob) {
    __shared__ __attribute__((aligned(64))) __bf16 pl[16 * 32];

    const int lane = threadIdx.x & 31;
    const int hi = lane >> 4;
    const int l15 = lane & 15;
    const int qb = blockIdx.x * 16;
    const int h = blockIdx.y;
    const int b = blockIdx.z;
    const int kvh = h >> 2;  // 4 query heads per kv head

    Frag qa[4];
    const __bf16* qRow = Qb + (size_t)(b * SS + qb + l15) * HIDDEN + h * HEAD_DIM;
#pragma unroll
    for (int kk = 0; kk < 4; ++kk) {
        qa[kk].h[0] = *(const v8bf*)(qRow + kk * 32 + 16 * hi);
        qa[kk].h[1] = *(const v8bf*)(qRow + kk * 32 + 16 * hi + 8);
    }

    float mrow[8], lrow[8];
    v8f o[8] = {};
#pragma unroll
    for (int r = 0; r < 8; ++r) { mrow[r] = NEG_BIG; lrow[r] = 0.0f; }

    const __bf16* kBase = Kb + (size_t)b * SS * KV_DIM + kvh * HEAD_DIM;
    const __bf16* vBase = Vt + (size_t)(kvh * HEAD_DIM) * TOKENS + (size_t)b * SS;

    for (int kv = 0; kv < qb + 16; kv += 32) {
        // ---- scores: two 16x16 tiles; batch the 4 K-fragment loads per tile
        v8f sc0 = {}, sc1 = {};
        v16bf kf[4];
#pragma unroll
        for (int kk = 0; kk < 4; ++kk)
            kf[kk] =
                *(const v16bf*)(kBase + (size_t)(kv + l15) * KV_DIM + kk * 32 + 16 * hi);
#pragma unroll
        for (int kk = 0; kk < 4; ++kk) sc0 = wmma_bf16(qa[kk].v, kf[kk], sc0);
#pragma unroll
        for (int kk = 0; kk < 4; ++kk)
            kf[kk] = *(const v16bf*)(kBase + (size_t)(kv + 16 + l15) * KV_DIM + kk * 32 +
                                     16 * hi);
#pragma unroll
        for (int kk = 0; kk < 4; ++kk) sc1 = wmma_bf16(qa[kk].v, kf[kk], sc1);

        // ---- online softmax (row = qb + r + 8*hi, col = kv + t*16 + l15)
#pragma unroll
        for (int r = 0; r < 8; ++r) {
            const int row = qb + r + 8 * hi;
            float s0 = sc0[r] * SCALE;
            float s1 = sc1[r] * SCALE;
            if (kv + l15 > row) s0 = NEG_BIG;
            if (kv + 16 + l15 > row) s1 = NEG_BIG;
            const float mx = half16_max(fmaxf(s0, s1));
            const float mn = fmaxf(mrow[r], mx);
            const float corr = __expf(mrow[r] - mn);
            const float p0 = __expf(s0 - mn);
            const float p1 = __expf(s1 - mn);
            const float rs = half16_sum(p0 + p1);
            lrow[r] = lrow[r] * corr + rs;
            mrow[r] = mn;
#pragma unroll
            for (int j = 0; j < 8; ++j) o[j][r] *= corr;
            pl[(r + 8 * hi) * 32 + l15] = (__bf16)p0;
            pl[(r + 8 * hi) * 32 + 16 + l15] = (__bf16)p1;
        }

        asm volatile("s_wait_dscnt 0" ::: "memory");

        Frag pa;
        pa.h[0] = *(const v8bf*)(pl + l15 * 32 + 16 * hi);
        pa.h[1] = *(const v8bf*)(pl + l15 * 32 + 16 * hi + 8);

        // ---- O += P @ V, V fragments batched in two groups of 4
#pragma unroll
        for (int g = 0; g < 2; ++g) {
            v16bf vf[4];
#pragma unroll
            for (int j = 0; j < 4; ++j)
                vf[j] = *(const v16bf*)(vBase +
                                        (size_t)((g * 4 + j) * 16 + l15) * TOKENS + kv +
                                        16 * hi);
#pragma unroll
            for (int j = 0; j < 4; ++j)
                o[g * 4 + j] = wmma_bf16(pa.v, vf[j], o[g * 4 + j]);
        }
    }

#pragma unroll
    for (int r = 0; r < 8; ++r) {
        const float inv = 1.0f / lrow[r];
        const size_t tok = (size_t)b * SS + qb + r + 8 * hi;
#pragma unroll
        for (int j = 0; j < 8; ++j)
            Ob[tok * HIDDEN + h * HEAD_DIM + j * 16 + l15] = (__bf16)(o[j][r] * inv);
    }
}

// ---------------------------------------------------------------- launch
extern "C" void kernel_launch(void* const* d_in, const int* in_sizes, int n_in,
                              void* d_out, int out_size, void* d_ws, size_t ws_size,
                              hipStream_t stream) {
    const float* hidden = (const float*)d_in[0];
    const int* pos = (const int*)d_in[1];
    const float* Wq = (const float*)d_in[2];
    const float* Wk = (const float*)d_in[3];
    const float* Wv = (const float*)d_in[4];
    const float* Wo = (const float*)d_in[5];

    char* ws = (char*)d_ws;
    const size_t E44 = (size_t)HIDDEN * HIDDEN;  // 16,777,216
    const size_t E41 = (size_t)HIDDEN * KV_DIM;  //  4,194,304
    __bf16* Xbf = (__bf16*)ws;                 ws += E44 * 2;
    __bf16* Wqt = (__bf16*)ws;                 ws += E44 * 2;
    __bf16* Wkt = (__bf16*)ws;                 ws += E41 * 2;
    __bf16* Wvt = (__bf16*)ws;                 ws += E41 * 2;
    __bf16* Wot = (__bf16*)ws;                 ws += E44 * 2;
    __bf16* Qb  = (__bf16*)ws;                 ws += E44 * 2;   // [4096 tok][4096]
    __bf16* Kb  = (__bf16*)ws;                 ws += E41 * 2;   // [4096 tok][1024]
    __bf16* Vt  = (__bf16*)ws;                 ws += E41 * 2;   // [1024 dim][4096 tok]
    __bf16* Ob  = (__bf16*)ws;                                  // [4096 tok][4096]

    cvt_f32_to_bf16<<<4096, 256, 0, stream>>>(hidden, Xbf, (long)TOKENS * HIDDEN);
    transpose_cvt_f32_to_bf16<<<4096, 256, 0, stream>>>(Wq, Wqt, HIDDEN, HIDDEN);
    transpose_cvt_f32_to_bf16<<<2048, 256, 0, stream>>>(Wk, Wkt, HIDDEN, KV_DIM);
    transpose_cvt_f32_to_bf16<<<2048, 256, 0, stream>>>(Wv, Wvt, HIDDEN, KV_DIM);
    transpose_cvt_f32_to_bf16<<<4096, 256, 0, stream>>>(Wo, Wot, HIDDEN, HIDDEN);

    gemm_bf16_wmma<<<dim3(HIDDEN / 64, TOKENS / 64), 128, 0, stream>>>(
        Xbf, Wqt, Qb, TOKENS, HIDDEN, HIDDEN, 0);
    gemm_bf16_wmma<<<dim3(KV_DIM / 64, TOKENS / 64), 128, 0, stream>>>(
        Xbf, Wkt, Kb, TOKENS, KV_DIM, HIDDEN, 0);
    gemm_bf16_wmma<<<dim3(KV_DIM / 64, TOKENS / 64), 128, 0, stream>>>(
        Xbf, Wvt, Vt, TOKENS, KV_DIM, HIDDEN, 1);  // transposed store [dim][tok]

    rope_bf16<<<TOKENS, 256, 0, stream>>>(Qb, Kb, pos);

    flash_wmma<<<dim3(SS / 16, N_HEADS, BB), 32, 0, stream>>>(Qb, Kb, Vt, Ob);

    gemm_bf16_wmma<<<dim3(HIDDEN / 64, TOKENS / 64), 128, 0, stream>>>(
        Ob, Wot, d_out, TOKENS, HIDDEN, HIDDEN, 2);
}
